// Attention_21449066676457
// MI455X (gfx1250) — matrix operand order repaired and checked by reading, not compiled, not run on hardware
//
#include <hip/hip_runtime.h>
#include <hip/hip_bf16.h>
#include <math.h>

// ---------------------------------------------------------------------------
// GQA attention block for MI455X (gfx1250, wave32, WMMA).
// f16 storage / f32 accumulation, v_wmma_f32_16x16x32_f16 for all matmuls.
// Global->LDS staging uses CDNA5 async loads (ASYNCcnt) with double buffering.
// Softmax reductions via DPP row_ror; exponentials via raw v_exp_f32.
// ---------------------------------------------------------------------------

typedef __attribute__((ext_vector_type(16))) _Float16 v16h;
typedef __attribute__((ext_vector_type(8)))  _Float16 v8h;
typedef __attribute__((ext_vector_type(4)))  _Float16 v4h;
typedef __attribute__((ext_vector_type(8)))  float    v8f;

namespace {
constexpr int B_   = 2;
constexpr int T_   = 2048;
constexpr int DM_  = 2048;
constexpr int H_   = 16;
constexpr int KV_  = 4;
constexpr int D_   = 128;
constexpr int NREP_ = H_ / KV_;
}

__device__ __forceinline__ v16h make_v16(v8h lo, v8h hi) {
  return __builtin_shufflevector(lo, hi, 0, 1, 2, 3, 4, 5, 6, 7,
                                         8, 9, 10, 11, 12, 13, 14, 15);
}

// A-matrix fragment (16x32 f16) from row-major LDS [m][k], stride ldk halves.
// lanes 0-15: M=lane, K={0..7,16..23}; lanes 16-31: M=lane-16, K={8..15,24..31}
__device__ __forceinline__ v16h load_frag_a(const _Float16* p, int ldk, int lane) {
  int m  = lane & 15;
  int kb = (lane < 16) ? 0 : 8;
  const v8h* q0 = reinterpret_cast<const v8h*>(p + m * ldk + kb);
  const v8h* q1 = reinterpret_cast<const v8h*>(p + m * ldk + kb + 16);
  return make_v16(*q0, *q1);
}

// B-matrix fragment (32x16 f16) from transposed LDS [n][k], stride ldk halves.
// lanes 0-15: N=lane, K={0..7,16..23}; lanes 16-31: N=lane-16, K={8..15,24..31}
__device__ __forceinline__ v16h load_frag_bt(const _Float16* p, int ldk, int lane) {
  int n  = lane & 15;
  int kb = (lane < 16) ? 0 : 8;
  const v8h* q0 = reinterpret_cast<const v8h*>(p + n * ldk + kb);
  const v8h* q1 = reinterpret_cast<const v8h*>(p + n * ldk + kb + 16);
  return make_v16(*q0, *q1);
}

// CDNA5 async global->LDS (tracked by ASYNCcnt). GVS mode: SGPR base + VGPR off.
__device__ __forceinline__ void async_ld_b128(uint32_t lds_byte_off,
                                              uint32_t gvoff_bytes,
                                              const void* base) {
  asm volatile("global_load_async_to_lds_b128 %0, %1, %2"
               :: "v"(lds_byte_off), "v"(gvoff_bytes), "s"(base)
               : "memory");
}
__device__ __forceinline__ void wait_async0() {
  asm volatile("s_wait_asynccnt 0x0" ::: "memory");
}
__device__ __forceinline__ uint32_t lds_off(const void* p) {
  return (uint32_t)(size_t)p;   // LDS aperture: addr[31:0] = LDS byte offset
}

// Raw v_exp_f32 (base-2). Flash-attn args are <= 0; hugely negative masked
// scores flush to 0 natively -- no range fixup sequence needed.
__device__ __forceinline__ float fast_exp2(float x) {
  return __builtin_amdgcn_exp2f(x);
}

// DPP row_ror reductions within each 16-lane half (pure VALU, no LDS traffic).
template <int CTRL>
__device__ __forceinline__ float dpp_mov(float x) {
  return __builtin_bit_cast(
      float, __builtin_amdgcn_update_dpp(0, __builtin_bit_cast(int, x), CTRL,
                                         0xf, 0xf, true));
}
__device__ __forceinline__ float half16_max(float v) {
  v = fmaxf(v, dpp_mov<0x121>(v));   // row_ror:1
  v = fmaxf(v, dpp_mov<0x122>(v));   // row_ror:2
  v = fmaxf(v, dpp_mov<0x124>(v));   // row_ror:4
  v = fmaxf(v, dpp_mov<0x128>(v));   // row_ror:8
  return v;
}
__device__ __forceinline__ float half16_sum(float v) {
  v += dpp_mov<0x121>(v);
  v += dpp_mov<0x122>(v);
  v += dpp_mov<0x124>(v);
  v += dpp_mov<0x128>(v);
  return v;
}

// ---------------------------------------------------------------------------
// fp32 -> f16 conversion (vectorized)
// ---------------------------------------------------------------------------
__global__ void cvt_f32_f16_kernel(const float* __restrict__ src,
                                   _Float16* __restrict__ dst, size_t n) {
  size_t i = ((size_t)blockIdx.x * blockDim.x + threadIdx.x) * 4;
  if (i + 3 < n) {
    float4 v = *reinterpret_cast<const float4*>(src + i);
    v4h h = { (_Float16)v.x, (_Float16)v.y, (_Float16)v.z, (_Float16)v.w };
    *reinterpret_cast<v4h*>(dst + i) = h;
  }
}

// ---------------------------------------------------------------------------
// WMMA GEMM: C[M,N] = A[M,K] @ B[K,N], row-major, f16 in, f16 or f32 out.
// Block tile 128x128, BK=32; 8 waves as 4(M) x 2(N); wave tile 32x64 (2x4 WMMA)
// A tile: async global->LDS, double buffered. B tile: register-prefetched one
// iteration ahead, scattered transposed into LDS before the barrier.
// ---------------------------------------------------------------------------
template <bool HALF_OUT>
__global__ __launch_bounds__(256) void gemm_f16_wmma(
    const _Float16* __restrict__ A, const _Float16* __restrict__ Bm,
    void* __restrict__ Cout, int M, int N, int K) {
  constexpr int BM = 128, BN = 128, BK = 32;
  constexpr int LDA = BK + 8;   // 80B rows keep 16B alignment
  constexpr int LDB = BK + 8;   // sBt is [n][k]
  __shared__ alignas(16) _Float16 sA[2][BM * LDA];
  __shared__ alignas(16) _Float16 sBt[2][BN * LDB];

  const int tid  = threadIdx.x;
  const int lane = tid & 31;
  const int wave = tid >> 5;     // 0..7
  const int wm   = wave >> 1;    // 0..3 -> M offset 32*wm
  const int wn   = wave & 1;     // 0..1 -> N offset 64*wn
  const int m0   = blockIdx.y * BM;
  const int n0   = blockIdx.x * BN;

  auto issueA = [&](int k0, int buf) {
#pragma unroll
    for (int it = 0; it < 2; ++it) {
      int linear = (it * 256 + tid) * 8;     // BM*BK = 4096 halves
      int r = linear / BK, c = linear % BK;
      uint32_t goff = (uint32_t)(((size_t)(m0 + r) * K + k0 + c) * 2);
      async_ld_b128(lds_off(&sA[buf][r * LDA + c]), goff, A);
    }
  };
  auto loadB = [&](int k0, v8h g[2]) {
#pragma unroll
    for (int it = 0; it < 2; ++it) {
      int linear = (it * 256 + tid) * 8;     // BK*BN = 4096 halves
      int r = linear / BN, c = linear % BN;
      g[it] = *reinterpret_cast<const v8h*>(Bm + (size_t)(k0 + r) * N + n0 + c);
    }
  };

  v8f acc[2][4] = {};
  v8h gB[2];
  issueA(0, 0);
  loadB(0, gB);

  const int KT = K / BK;
  for (int kt = 0; kt < KT; ++kt) {
    const int cur = kt & 1, nxt = cur ^ 1;
    wait_async0();                          // A tile kt resident (own share)
    // Scatter prefetched B tile (kt) transposed into sBt[cur].
#pragma unroll
    for (int it = 0; it < 2; ++it) {
      int linear = (it * 256 + tid) * 8;
      int r = linear / BN, c = linear % BN;
#pragma unroll
      for (int j = 0; j < 8; ++j) sBt[cur][(c + j) * LDB + r] = gB[it][j];
    }
    __syncthreads();                        // tiles kt visible; kt-1 compute done
    if (kt + 1 < KT) {
      issueA((kt + 1) * BK, nxt);           // safe: all waves past compute(kt-1)
      loadB((kt + 1) * BK, gB);
      if (kt + 2 < KT) {                    // warm L2 two tiles ahead
        int linear = tid * 8;
        __builtin_prefetch(
            Bm + (size_t)((kt + 2) * BK + linear / BN) * N + n0 + linear % BN,
            0, 1);
        __builtin_prefetch(
            A + (size_t)(m0 + linear / BK) * K + (kt + 2) * BK + linear % BK,
            0, 1);
      }
    }
#pragma unroll
    for (int i = 0; i < 2; ++i) {
      v16h af = load_frag_a(&sA[cur][(wm * 32 + i * 16) * LDA], LDA, lane);
#pragma unroll
      for (int j = 0; j < 4; ++j) {
        v16h bf = load_frag_bt(&sBt[cur][(wn * 64 + j * 16) * LDB], LDB, lane);
        acc[i][j] = __builtin_amdgcn_wmma_f32_16x16x32_f16(
            false, af, false, bf, (short)0, acc[i][j], false, false);
      }
    }
  }

  // C/D layout: VGPR r holds rows {r, r+8}; lanes 0-15 / 16-31 are the halves.
#pragma unroll
  for (int i = 0; i < 2; ++i) {
    int mbase = m0 + wm * 32 + i * 16 + ((lane < 16) ? 0 : 8);
#pragma unroll
    for (int j = 0; j < 4; ++j) {
      int n = n0 + wn * 64 + j * 16 + (lane & 15);
#pragma unroll
      for (int r = 0; r < 8; ++r) {
        if (HALF_OUT)
          ((_Float16*)Cout)[(size_t)(mbase + r) * N + n] = (_Float16)acc[i][j][r];
        else
          ((float*)Cout)[(size_t)(mbase + r) * N + n] = acc[i][j][r];
      }
    }
  }
}

// ---------------------------------------------------------------------------
// RoPE in-place on f16 [B, T, heads*D] (rotate-half convention).
// ---------------------------------------------------------------------------
__global__ void rope_kernel(_Float16* __restrict__ X, int heads) {
  size_t idx = (size_t)blockIdx.x * blockDim.x + threadIdx.x;
  size_t total = (size_t)B_ * T_ * heads * (D_ / 2);
  if (idx >= total) return;
  int i = (int)(idx % (D_ / 2));
  size_t tmp = idx / (D_ / 2);
  int hh = (int)(tmp % heads);
  tmp /= heads;
  int t = (int)(tmp % T_);
  int b = (int)(tmp / T_);
  // inv_freq = 10000^(-2i/128) = exp(-i * ln(10000)/64)
  float theta = (float)t * __expf(-(float)i * 0.14391156831212787f);
  float s, c;
  __sincosf(theta, &s, &c);
  _Float16* base = X + (((size_t)b * T_ + t) * heads + hh) * D_;
  float x0 = (float)base[i], x1 = (float)base[i + 64];
  base[i]      = (_Float16)(x0 * c - x1 * s);
  base[i + 64] = (_Float16)(x1 * c + x0 * s);
}

// ---------------------------------------------------------------------------
// Causal GQA flash attention. Grid: (T/128, H, B); 8 waves x 16 q-rows.
// K tiles: async global->LDS, double buffered. V tiles: register-prefetched,
// scattered transposed (two-barrier scheme). Softmax reductions via DPP.
// ---------------------------------------------------------------------------
__global__ __launch_bounds__(256) void flash_attn_wmma(
    const _Float16* __restrict__ Q, const _Float16* __restrict__ Km,
    const _Float16* __restrict__ Vm, _Float16* __restrict__ O) {
  constexpr int BQ  = 128;
  constexpr int BKV = 64;
  constexpr int LDK = D_ + 8;    // sK  [kcol][d] -> transposed-B for QK^T
  constexpr int LDV = BKV + 8;   // sVt [d][kcol] -> transposed-B for P@V
  constexpr int LDP = 32 + 8;    // per-wave P staging, 16x32 per pass
  __shared__ alignas(16) _Float16 sK[2][BKV * LDK];
  __shared__ alignas(16) _Float16 sVt[D_ * LDV];
  __shared__ alignas(16) _Float16 sP[8 * 16 * LDP];

  const int tid  = threadIdx.x;
  const int lane = tid & 31;
  const int wave = tid >> 5;
  const int q0   = blockIdx.x * BQ;
  const int h    = blockIdx.y;
  const int b    = blockIdx.z;
  const int kvh  = h / NREP_;

  const float scale = 0.08838834764831845f;   // 1/sqrt(128)
  const float log2e = 1.4426950408889634f;

  auto issueK = [&](int kb, int buf) {
#pragma unroll
    for (int it = 0; it < 4; ++it) {
      int linear = (it * 256 + tid) * 8;     // BKV*D = 8192 halves
      int r = linear / D_, c = linear % D_;
      size_t goff =
          (((size_t)b * T_ + kb * BKV + r) * (KV_ * D_) + kvh * D_ + c) * 2;
      async_ld_b128(lds_off(&sK[buf][r * LDK + c]), (uint32_t)goff, Km);
    }
  };
  auto loadV = [&](int kb, v8h g[4]) {
#pragma unroll
    for (int it = 0; it < 4; ++it) {
      int linear = (it * 256 + tid) * 8;
      int r = linear / D_, c = linear % D_;
      g[it] = *reinterpret_cast<const v8h*>(
          Vm + ((size_t)b * T_ + kb * BKV + r) * (KV_ * D_) + kvh * D_ + c);
    }
  };

  // Q fragments for this wave's 16 rows: D=128 -> 4 K-steps of 32, from global.
  v16h qf[4];
  {
    int m  = lane & 15;
    int kb = (lane < 16) ? 0 : 8;
    const _Float16* qrow =
        Q + ((size_t)b * T_ + q0 + wave * 16 + m) * (H_ * D_) + h * D_;
#pragma unroll
    for (int s = 0; s < 4; ++s) {
      const v8h* p0 = reinterpret_cast<const v8h*>(qrow + s * 32 + kb);
      const v8h* p1 = reinterpret_cast<const v8h*>(qrow + s * 32 + kb + 16);
      qf[s] = make_v16(*p0, *p1);
    }
  }

  v8f o[8] = {};                 // 16 x 128 output accumulator (8 N-tiles)
  float mrow[8], lrow[8];
#pragma unroll
  for (int r = 0; r < 8; ++r) { mrow[r] = -1e30f; lrow[r] = 0.f; }

  const int qrow_base = q0 + wave * 16;
  const int nkb = (q0 + BQ) / BKV;            // causal upper bound

  v8h gv[4];
  issueK(0, 0);
  loadV(0, gv);

  for (int kb = 0; kb < nkb; ++kb) {
    const int cur = kb & 1, nxt = cur ^ 1;
    wait_async0();               // K tile kb resident (own share)
    __syncthreads();             // barrier 1: K visible; compute(kb-1) done
    if (kb + 1 < nkb) {
      issueK(kb + 1, nxt);       // safe: sK[nxt] readers all finished
    }
    // Scatter prefetched V tile (kb) transposed into sVt (freed by barrier 1).
#pragma unroll
    for (int it = 0; it < 4; ++it) {
      int linear = (it * 256 + tid) * 8;
      int r = linear / D_, c = linear % D_;
#pragma unroll
      for (int j = 0; j < 8; ++j) sVt[(c + j) * LDV + r] = gv[it][j];
    }
    if (kb + 1 < nkb) {
      loadV(kb + 1, gv);
      if (kb + 2 < nkb) {        // warm L2 two tiles ahead
        size_t goff =
            ((size_t)b * T_ + (kb + 2) * BKV + (tid >> 4)) * (KV_ * D_) +
            kvh * D_ + (tid & 15) * 8;
        __builtin_prefetch(Km + goff, 0, 1);
        __builtin_prefetch(Vm + goff, 0, 1);
      }
    }

    // S(16x64) = Q @ K^T : 4 N-tiles x 4 K-steps (overlaps V scatter latency).
    v8f S[4];
#pragma unroll
    for (int t = 0; t < 4; ++t) {
      v8f a = {};
#pragma unroll
      for (int s = 0; s < 4; ++s) {
        v16h bf = load_frag_bt(&sK[cur][(t * 16) * LDK + s * 32], LDK, lane);
        a = __builtin_amdgcn_wmma_f32_16x16x32_f16(false, qf[s], false, bf,
                                                   (short)0, a, false, false);
      }
#pragma unroll
      for (int r = 0; r < 8; ++r) a[r] *= scale;
      S[t] = a;
    }

    // Causal mask (analytic; only diagonal-touching tiles need it).
    if (kb * BKV + BKV - 1 > qrow_base) {
#pragma unroll
      for (int t = 0; t < 4; ++t) {
        int kcol = kb * BKV + t * 16 + (lane & 15);
#pragma unroll
        for (int r = 0; r < 8; ++r) {
          int qrow = qrow_base + r + ((lane < 16) ? 0 : 8);
          if (kcol > qrow) S[t][r] = -1e30f;
        }
      }
    }

    // Online softmax: per-row reductions across the owning 16-lane half (DPP);
    // exponentials are single v_exp_f32 (TRANS -> co-executes with WMMA).
#pragma unroll
    for (int r = 0; r < 8; ++r) {
      float v = fmaxf(fmaxf(S[0][r], S[1][r]), fmaxf(S[2][r], S[3][r]));
      v = half16_max(v);
      float mnew  = fmaxf(mrow[r], v);
      float alpha = fast_exp2((mrow[r] - mnew) * log2e);
      mrow[r] = mnew;
      lrow[r] *= alpha;
#pragma unroll
      for (int t = 0; t < 8; ++t) o[t][r] *= alpha;
      float psum = 0.f;
#pragma unroll
      for (int t = 0; t < 4; ++t) {
        float p = fast_exp2((S[t][r] - mnew) * log2e);
        S[t][r] = p;
        psum += p;
      }
      lrow[r] += half16_sum(psum);
    }

    __syncthreads();             // barrier 2: sVt writes visible to all waves

    // o(16x128) += P(16x64) @ V(64x128): two 16x32 passes through per-wave LDS
    // (C-layout -> A-layout reshape), 8 N-tiles each.
    _Float16* pP = &sP[wave * 16 * LDP];
#pragma unroll
    for (int s = 0; s < 2; ++s) {
#pragma unroll
      for (int t2 = 0; t2 < 2; ++t2) {
        int n  = t2 * 16 + (lane & 15);
        int rb = (lane < 16) ? 0 : 8;
#pragma unroll
        for (int r = 0; r < 8; ++r)
          pP[(rb + r) * LDP + n] = (_Float16)S[2 * s + t2][r];
      }
      asm volatile("s_wait_dscnt 0x0" ::: "memory");  // same-wave LDS RAW
      v16h pf = load_frag_a(pP, LDP, lane);
#pragma unroll
      for (int t = 0; t < 8; ++t) {
        v16h vf = load_frag_bt(&sVt[(t * 16) * LDV + s * 32], LDV, lane);
        o[t] = __builtin_amdgcn_wmma_f32_16x16x32_f16(false, pf, false, vf,
                                                      (short)0, o[t], false,
                                                      false);
      }
    }
  }

  // Epilogue: normalize by l, store f16 to [b, q, h*D + d] for the Wo GEMM.
#pragma unroll
  for (int t = 0; t < 8; ++t) {
    int n  = t * 16 + (lane & 15);
    int rb = (lane < 16) ? 0 : 8;
#pragma unroll
    for (int r = 0; r < 8; ++r) {
      int qrow  = q0 + wave * 16 + rb + r;
      float inv = 1.f / lrow[r];
      O[((size_t)b * T_ + qrow) * (H_ * D_) + h * D_ + n] =
          (_Float16)(o[t][r] * inv);
    }
  }
}

// ---------------------------------------------------------------------------
// Launch: x,Wq,Wk,Wv,Wo are fp32 inputs; mask input ignored (causal analytic).
// ---------------------------------------------------------------------------
extern "C" void kernel_launch(void* const* d_in, const int* in_sizes, int n_in,
                              void* d_out, int out_size, void* d_ws,
                              size_t ws_size, hipStream_t stream) {
  (void)in_sizes; (void)n_in; (void)out_size; (void)ws_size;
  const float* x  = (const float*)d_in[0];
  const float* Wq = (const float*)d_in[1];
  const float* Wk = (const float*)d_in[2];
  const float* Wv = (const float*)d_in[3];
  const float* Wo = (const float*)d_in[4];

  char* ws = (char*)d_ws;
  size_t off = 0;
  auto alloc = [&](size_t bytes) -> char* {
    char* p = ws + off;
    off += (bytes + 255) & ~(size_t)255;
    return p;
  };
  _Float16* xh  = (_Float16*)alloc((size_t)B_ * T_ * DM_ * 2);
  _Float16* Wqh = (_Float16*)alloc((size_t)DM_ * H_ * D_ * 2);
  _Float16* Wkh = (_Float16*)alloc((size_t)DM_ * KV_ * D_ * 2);
  _Float16* Wvh = (_Float16*)alloc((size_t)DM_ * KV_ * D_ * 2);
  _Float16* Woh = (_Float16*)alloc((size_t)H_ * D_ * DM_ * 2);
  _Float16* Qh  = (_Float16*)alloc((size_t)B_ * T_ * H_ * D_ * 2);
  _Float16* Kh  = (_Float16*)alloc((size_t)B_ * T_ * KV_ * D_ * 2);
  _Float16* Vh  = (_Float16*)alloc((size_t)B_ * T_ * KV_ * D_ * 2);
  _Float16* Ah  = (_Float16*)alloc((size_t)B_ * T_ * H_ * D_ * 2);

  auto cvt = [&](const float* s, _Float16* d, size_t n) {
    cvt_f32_f16_kernel<<<dim3((unsigned)((n / 4 + 255) / 256)), dim3(256), 0,
                         stream>>>(s, d, n);
  };
  cvt(x,  xh,  (size_t)B_ * T_ * DM_);
  cvt(Wq, Wqh, (size_t)DM_ * H_ * D_);
  cvt(Wk, Wkh, (size_t)DM_ * KV_ * D_);
  cvt(Wv, Wvh, (size_t)DM_ * KV_ * D_);
  cvt(Wo, Woh, (size_t)H_ * D_ * DM_);

  const int M = B_ * T_;
  gemm_f16_wmma<true><<<dim3((H_ * D_) / 128, M / 128), 256, 0, stream>>>(
      xh, Wqh, Qh, M, H_ * D_, DM_);
  gemm_f16_wmma<true><<<dim3((KV_ * D_) / 128, M / 128), 256, 0, stream>>>(
      xh, Wkh, Kh, M, KV_ * D_, DM_);
  gemm_f16_wmma<true><<<dim3((KV_ * D_) / 128, M / 128), 256, 0, stream>>>(
      xh, Wvh, Vh, M, KV_ * D_, DM_);

  size_t nq = (size_t)B_ * T_ * H_ * (D_ / 2);
  rope_kernel<<<dim3((unsigned)((nq + 255) / 256)), 256, 0, stream>>>(Qh, H_);
  size_t nk = (size_t)B_ * T_ * KV_ * (D_ / 2);
  rope_kernel<<<dim3((unsigned)((nk + 255) / 256)), 256, 0, stream>>>(Kh, KV_);

  flash_attn_wmma<<<dim3(T_ / 128, H_, B_), 256, 0, stream>>>(Qh, Kh, Vh, Ah);

  gemm_f16_wmma<false><<<dim3(DM_ / 128, M / 128), 256, 0, stream>>>(
      Ah, Woh, (float*)d_out, M, DM_, DM_);
}
